// MultiHeadSA_74775380623733
// MI455X (gfx1250) — matrix operand order repaired and checked by reading, not compiled, not run on hardware
//
#include <hip/hip_runtime.h>
#include <cstdint>
#include <cstddef>

// ---------------------------------------------------------------------------
// MultiHeadSA for gfx1250 (CDNA5). All GEMMs via v_wmma_f32_16x16x32_bf16.
// Pipeline:
//   1) convert inputs f32[N,D,P] -> bf16 inT[N,P,D] (transposed)
//   2) convert Wk/Wq/Wv/Wo f32 -> bf16 (natural row-major, contiguous K)
//   3) K,Q projection GEMMs  -> bf16 [N,H,P,D]
//      V projection GEMM     -> bf16 [N,H,D,P]
//   4) fused attention: per (n,h,q16) workgroup (4 waves):
//        S = K^T Q (scores in VGPRs), +pos/sqrt(D), softmax over k (key axis),
//        probs -> LDS bf16 [16][1032], O = V*P -> bf16 out[N,P,H*D]
//   5) output projection y = Wo*out + bo -> f32 [N,D,P]
// All operand loads are depth-1 software-pipelined so WMMAs never wait on
// loads issued in the same iteration.
// ---------------------------------------------------------------------------

typedef __bf16 bf16_t;
typedef __attribute__((ext_vector_type(16))) __bf16        v16bf;
typedef __attribute__((ext_vector_type(8)))  float         v8f;
typedef __attribute__((ext_vector_type(4)))  unsigned int  u32x4;

#define NB 8      // batch
#define HH 8      // heads
#define DD 256    // head dim
#define PP 1024   // patches
#define HD 2048   // H*D

// ----------------------------- helpers -------------------------------------

static __device__ __forceinline__ unsigned int bf16_1(float f) {
  unsigned int u = __builtin_bit_cast(unsigned int, f);
  // round-to-nearest-even f32 -> bf16
  return (u + 0x7FFFu + ((u >> 16) & 1u)) >> 16;
}
static __device__ __forceinline__ unsigned int pack2(float lo, float hi) {
  return bf16_1(lo) | (bf16_1(hi) << 16);
}

union V16U { u32x4 u[2]; v16bf v; };

// A operand (16x32, bf16), memory row-major [M][K], ld in elements.
// lane L<16: row L, K {k0..k0+7} and {k0+16..k0+23}; lane L>=16: +8 offset.
static __device__ __forceinline__ v16bf load_a16(const unsigned short* base,
                                                 int ld, int k0, int lane) {
  int m = lane & 15, hi = lane >> 4;
  const unsigned short* p = base + (size_t)m * ld + k0 + hi * 8;
  V16U t;
  t.u[0] = *(const u32x4*)(p);
  t.u[1] = *(const u32x4*)(p + 16);
  return t.v;
}

// B operand (32x16, bf16), memory as [N][K] (transposed row-major), ld in elems.
// lane L<16: col L, K {k0..k0+7} and {k0+16..k0+23}; lane L>=16: +8 offset.
static __device__ __forceinline__ v16bf load_b16(const unsigned short* base,
                                                 int ld, int k0, int lane) {
  int c = lane & 15, hi = lane >> 4;
  const unsigned short* p = base + (size_t)c * ld + k0 + hi * 8;
  V16U t;
  t.u[0] = *(const u32x4*)(p);
  t.u[1] = *(const u32x4*)(p + 16);
  return t.v;
}

static __device__ __forceinline__ v8f wmma_bf16(v16bf a, v16bf b, v8f c) {
  return __builtin_amdgcn_wmma_f32_16x16x32_bf16(false, a, false, b,
                                                 (short)0, c, false, false);
}

// ------------------------- conversion kernels ------------------------------

__global__ __launch_bounds__(256) void cvt_bf16_kernel(
    const float* __restrict__ src, unsigned short* __restrict__ dst, int count) {
  int i = blockIdx.x * 256 + threadIdx.x;
  if (i < count) dst[i] = (unsigned short)bf16_1(src[i]);
}

// inputs f32 [N][D][P] -> bf16 inT [N][P][D]
__global__ __launch_bounds__(256) void transpose_in_kernel(
    const float* __restrict__ src, unsigned short* __restrict__ dst) {
  int i = blockIdx.x * 256 + threadIdx.x;     // N*D*P = 2^21 total
  int p = i & (PP - 1);
  int d = (i >> 10) & (DD - 1);
  int n = i >> 18;
  dst[((size_t)n * PP + p) * DD + d] = (unsigned short)bf16_1(src[i]);
}

// --------------------------- K/Q projection --------------------------------
// C = W[2048,256] * inT^T  ->  stored [n][h][p][d] (contiguous d).
// 4 o-tiles per wave share the B (inT) operand; operands double-buffered.
__global__ __launch_bounds__(128) void proj_kq_kernel(
    const unsigned short* __restrict__ Wb, const float* __restrict__ bias,
    const unsigned short* __restrict__ inT, unsigned short* __restrict__ dst) {
  int wave = threadIdx.x >> 5, lane = threadIdx.x & 31;
  int gt = blockIdx.x * 4 + wave;           // n(8) * og(32) * pt(64) = 16384
  int pt = gt & 63;
  int og = (gt >> 6) & 31;
  int n  = gt >> 11;
  int p0 = pt * 16, o0 = og * 64;
  int lp = lane & 15, hi = lane >> 4;

  const unsigned short* B0 = inT + ((size_t)n * PP + p0) * DD;
  v8f c[4] = {v8f{}, v8f{}, v8f{}, v8f{}};

  v16bf bC = load_b16(B0, DD, 0, lane);
  v16bf aC[4];
#pragma unroll
  for (int t = 0; t < 4; ++t)
    aC[t] = load_a16(Wb + (size_t)(o0 + t * 16) * DD, DD, 0, lane);

#pragma unroll
  for (int ch = 0; ch < 8; ++ch) {
    int k1 = ((ch + 1) & 7) * 32;
    v16bf bN = load_b16(B0, DD, k1, lane);
    v16bf aN[4];
#pragma unroll
    for (int t = 0; t < 4; ++t)
      aN[t] = load_a16(Wb + (size_t)(o0 + t * 16) * DD, DD, k1, lane);
#pragma unroll
    for (int t = 0; t < 4; ++t) c[t] = wmma_bf16(aC[t], bC, c[t]);
    bC = bN;
#pragma unroll
    for (int t = 0; t < 4; ++t) aC[t] = aN[t];
  }

#pragma unroll
  for (int t = 0; t < 4; ++t) {
    int ob = o0 + t * 16 + hi * 8;          // 8 consecutive o's per lane
    int h = ob >> 8, dl = ob & 255;
    unsigned short* out =
        dst + (((size_t)n * HH + h) * PP + p0 + lp) * DD + dl;
    u32x4 pk;
    pk.x = pack2(c[t][0] + bias[ob + 0], c[t][1] + bias[ob + 1]);
    pk.y = pack2(c[t][2] + bias[ob + 2], c[t][3] + bias[ob + 3]);
    pk.z = pack2(c[t][4] + bias[ob + 4], c[t][5] + bias[ob + 5]);
    pk.w = pack2(c[t][6] + bias[ob + 6], c[t][7] + bias[ob + 7]);
    *(u32x4*)out = pk;
  }
}

// ----------------------------- V projection --------------------------------
// C' = inT * Wv^T (M=p, N=o)  ->  stored [n][h][d][p] (contiguous p).
// 4 o-tiles per wave share the A (inT) operand; operands double-buffered.
__global__ __launch_bounds__(128) void proj_v_kernel(
    const unsigned short* __restrict__ Wb, const float* __restrict__ bias,
    const unsigned short* __restrict__ inT, unsigned short* __restrict__ dst) {
  int wave = threadIdx.x >> 5, lane = threadIdx.x & 31;
  int gt = blockIdx.x * 4 + wave;           // n(8) * og(32) * pt(64) = 16384
  int pt = gt & 63;
  int og = (gt >> 6) & 31;
  int n  = gt >> 11;
  int p0 = pt * 16, o0 = og * 64;
  int lo = lane & 15, hi = lane >> 4;

  const unsigned short* A0 = inT + ((size_t)n * PP + p0) * DD;
  v8f c[4] = {v8f{}, v8f{}, v8f{}, v8f{}};

  v16bf aC = load_a16(A0, DD, 0, lane);
  v16bf bC[4];
#pragma unroll
  for (int t = 0; t < 4; ++t)
    bC[t] = load_b16(Wb + (size_t)(o0 + t * 16) * DD, DD, 0, lane);

#pragma unroll
  for (int ch = 0; ch < 8; ++ch) {
    int k1 = ((ch + 1) & 7) * 32;
    v16bf aN = load_a16(A0, DD, k1, lane);
    v16bf bN[4];
#pragma unroll
    for (int t = 0; t < 4; ++t)
      bN[t] = load_b16(Wb + (size_t)(o0 + t * 16) * DD, DD, k1, lane);
#pragma unroll
    for (int t = 0; t < 4; ++t) c[t] = wmma_bf16(aC, bC[t], c[t]);
    aC = aN;
#pragma unroll
    for (int t = 0; t < 4; ++t) bC[t] = bN[t];
  }

#pragma unroll
  for (int t = 0; t < 4; ++t) {
    int o = o0 + t * 16 + lo;               // fixed o (=> fixed d row) per lane
    int h = o >> 8, dl = o & 255;
    float bi = bias[o];
    unsigned short* out =
        dst + (((size_t)n * HH + h) * DD + dl) * PP + p0 + hi * 8;
    u32x4 pk;
    pk.x = pack2(c[t][0] + bi, c[t][1] + bi);
    pk.y = pack2(c[t][2] + bi, c[t][3] + bi);
    pk.z = pack2(c[t][4] + bi, c[t][5] + bi);
    pk.w = pack2(c[t][6] + bi, c[t][7] + bi);
    *(u32x4*)out = pk;
  }
}

// ------------------------- fused attention ---------------------------------
// One workgroup (4 waves) per (n, h, 16-query tile).
// Wave w owns key range [w*256, w*256+256): 16 score tiles kept in VGPRs.
// Softmax over the key axis (reference softmax dim=2), probs to LDS bf16,
// then O = V * P with d split across the 4 waves.
__global__ __launch_bounds__(128) void attn_kernel(
    const unsigned short* __restrict__ K, const unsigned short* __restrict__ Q,
    const unsigned short* __restrict__ V, const float* __restrict__ pos,
    unsigned short* __restrict__ outT) {
  int bid = blockIdx.x;                     // n(8)*h(8)*qt(64) = 4096
  int qt = bid & 63;
  int h  = (bid >> 6) & 7;
  int n  = bid >> 9;
  int q0 = qt * 16;
  int wave = threadIdx.x >> 5, lane = threadIdx.x & 31;
  int lq = lane & 15, hi = lane >> 4;

  __shared__ __align__(16) unsigned short probs[16 * 1032];  // [q][k], padded
  __shared__ float red[2][4][16];                            // max/sum, wave, q

  const unsigned short* Kt = K + ((size_t)n * HH + h) * PP * DD;  // [p][d]
  const unsigned short* Qt = Q + ((size_t)n * HH + h) * PP * DD;  // [p][d]
  const unsigned short* Vm = V + ((size_t)n * HH + h) * DD * PP;  // [d][p]

  // B operand (Q columns q0..q0+15) for all 8 d-chunks, loaded once.
  v16bf Bq[8];
#pragma unroll
  for (int ch = 0; ch < 8; ++ch)
    Bq[ch] = load_b16(Qt + (size_t)q0 * DD, DD, ch * 32, lane);

  // ---- scores S = K^T Q for this wave's 256 keys (16 tiles in VGPRs) ----
  // A tiles double-buffered: load for iteration i+1 issued before WMMA i.
  int kbase = wave * 256;
  v8f acc[16];
#pragma unroll
  for (int kt = 0; kt < 16; ++kt) acc[kt] = v8f{};

  v16bf a_cur = load_a16(Kt + (size_t)kbase * DD, DD, 0, lane);
#pragma unroll
  for (int it = 0; it < 128; ++it) {
    int kt = it >> 3, ch = it & 7;
    int it2 = (it + 1) & 127;
    v16bf a_nxt = load_a16(Kt + (size_t)(kbase + (it2 >> 3) * 16) * DD, DD,
                           (it2 & 7) * 32, lane);
    acc[kt] = wmma_bf16(a_cur, Bq[ch], acc[kt]);
    a_cur = a_nxt;
  }

  // ---- scale + positional bias, running max over key axis ----
  const float scale = 0.0625f;  // 1/sqrt(256)
  const float* posq = pos + (size_t)h * PP * PP + q0 + lq;
  float m = -3.0e38f;
#pragma unroll
  for (int kt = 0; kt < 16; ++kt) {
    int kr = kbase + kt * 16 + hi * 8;
#pragma unroll
    for (int j = 0; j < 8; ++j) {
      float s = acc[kt][j] * scale + posq[(size_t)(kr + j) * PP];
      acc[kt][j] = s;
      m = fmaxf(m, s);
    }
  }
  // lanes L and L+16 hold the same q, different k rows
  m = fmaxf(m, __shfl_xor(m, 16, 32));
  if (lane < 16) red[0][wave][lq] = m;
  __syncthreads();
  float mall = fmaxf(fmaxf(red[0][0][lq], red[0][1][lq]),
                     fmaxf(red[0][2][lq], red[0][3][lq]));

  // ---- exp and sum over key axis ----
  float s = 0.0f;
#pragma unroll
  for (int kt = 0; kt < 16; ++kt) {
#pragma unroll
    for (int j = 0; j < 8; ++j) {
      float e = __expf(acc[kt][j] - mall);
      acc[kt][j] = e;
      s += e;
    }
  }
  s += __shfl_xor(s, 16, 32);
  if (lane < 16) red[1][wave][lq] = s;
  __syncthreads();
  float inv = 1.0f / (red[1][0][lq] + red[1][1][lq] +
                      red[1][2][lq] + red[1][3][lq]);

  // ---- probs (bf16) into LDS [q][1032], 16B packed per lane per tile ----
#pragma unroll
  for (int kt = 0; kt < 16; ++kt) {
    int kr = kbase + kt * 16 + hi * 8;      // 8 consecutive k per lane
    u32x4 pk;
    pk.x = pack2(acc[kt][0] * inv, acc[kt][1] * inv);
    pk.y = pack2(acc[kt][2] * inv, acc[kt][3] * inv);
    pk.z = pack2(acc[kt][4] * inv, acc[kt][5] * inv);
    pk.w = pack2(acc[kt][6] * inv, acc[kt][7] * inv);
    *(u32x4*)&probs[lq * 1032 + kr] = pk;
  }
  __syncthreads();

  // ---- O = V * P : wave owns 64 rows of d (4 C-tiles), double-buffered ----
  v8f o[4] = {v8f{}, v8f{}, v8f{}, v8f{}};
  int d0w = wave * 64;
  v16bf b_cur = load_b16(probs, 1032, 0, lane);
  v16bf v_cur = load_a16(Vm + (size_t)d0w * PP, PP, 0, lane);
#pragma unroll 4
  for (int it = 0; it < 128; ++it) {
    int t = it & 3;                          // constant under unroll-4
    int it2 = (it + 1) & 127;
    int kc2 = it2 >> 2, t2 = it2 & 3;
    v16bf v_nxt = load_a16(Vm + (size_t)(d0w + t2 * 16) * PP, PP, kc2 * 32, lane);
    v16bf b_nxt = b_cur;
    if (t == 3) b_nxt = load_b16(probs, 1032, kc2 * 32, lane);
    o[t] = wmma_bf16(v_cur, b_cur, o[t]);
    v_cur = v_nxt;
    b_cur = b_nxt;
  }

  // ---- store out[n][q][h*256 + d], 16B packed per lane per tile ----
  unsigned short* op = outT + ((size_t)n * PP + q0 + lq) * HD + h * DD;
#pragma unroll
  for (int t = 0; t < 4; ++t) {
    int d0 = d0w + t * 16 + hi * 8;
    u32x4 pk;
    pk.x = pack2(o[t][0], o[t][1]);
    pk.y = pack2(o[t][2], o[t][3]);
    pk.z = pack2(o[t][4], o[t][5]);
    pk.w = pack2(o[t][6], o[t][7]);
    *(u32x4*)(op + d0) = pk;
  }
}

// --------------------------- output projection -----------------------------
// y[n][o][p] = sum_i Wo[o][i] * out[n][p][i] + bo[o]   (f32 result)
__global__ __launch_bounds__(128) void out_proj_kernel(
    const unsigned short* __restrict__ Wob, const float* __restrict__ bo,
    const unsigned short* __restrict__ outT, float* __restrict__ y) {
  int wave = threadIdx.x >> 5, lane = threadIdx.x & 31;
  int gt = blockIdx.x * 4 + wave;           // n(8) * og(4) * pt(64) = 2048
  int pt = gt & 63;
  int og = (gt >> 6) & 3;
  int n  = gt >> 8;
  int p0 = pt * 16, o0 = og * 64;
  int lq = lane & 15, hi = lane >> 4;

  const unsigned short* B0 = outT + ((size_t)n * PP + p0) * HD;
  v8f c[4] = {v8f{}, v8f{}, v8f{}, v8f{}};

  v16bf bC = load_b16(B0, HD, 0, lane);
  v16bf aC[4];
#pragma unroll
  for (int t = 0; t < 4; ++t)
    aC[t] = load_a16(Wob + (size_t)(o0 + t * 16) * HD, HD, 0, lane);

#pragma unroll 2
  for (int ch = 0; ch < 64; ++ch) {
    int k1 = ((ch + 1) & 63) * 32;
    v16bf bN = load_b16(B0, HD, k1, lane);
    v16bf aN[4];
#pragma unroll
    for (int t = 0; t < 4; ++t)
      aN[t] = load_a16(Wob + (size_t)(o0 + t * 16) * HD, HD, k1, lane);
#pragma unroll
    for (int t = 0; t < 4; ++t) c[t] = wmma_bf16(aC[t], bC, c[t]);
    bC = bN;
#pragma unroll
    for (int t = 0; t < 4; ++t) aC[t] = aN[t];
  }

#pragma unroll
  for (int t = 0; t < 4; ++t) {
#pragma unroll
    for (int j = 0; j < 8; ++j) {
      int o = o0 + t * 16 + hi * 8 + j;
      y[((size_t)n * DD + o) * PP + p0 + lq] = c[t][j] + bo[o];
    }
  }
}

// ------------------------------- launch ------------------------------------

extern "C" void kernel_launch(void* const* d_in, const int* in_sizes, int n_in,
                              void* d_out, int out_size, void* d_ws,
                              size_t ws_size, hipStream_t stream) {
  const float* x   = (const float*)d_in[0];   // [N,D,P]
  const float* pos = (const float*)d_in[1];   // [1,H,P,P]
  const float* Wk  = (const float*)d_in[2];
  const float* bk  = (const float*)d_in[3];
  const float* Wq  = (const float*)d_in[4];
  const float* bq  = (const float*)d_in[5];
  const float* Wv  = (const float*)d_in[6];
  const float* bv  = (const float*)d_in[7];
  const float* Wo  = (const float*)d_in[8];
  const float* bo  = (const float*)d_in[9];

  char* ws = (char*)d_ws;
  const size_t MB = 1u << 20;
  unsigned short* inT = (unsigned short*)(ws + 0 * MB);    // 4 MB
  unsigned short* WkB = (unsigned short*)(ws + 4 * MB);    // 1 MB
  unsigned short* WqB = (unsigned short*)(ws + 5 * MB);    // 1 MB
  unsigned short* WvB = (unsigned short*)(ws + 6 * MB);    // 1 MB
  unsigned short* WoB = (unsigned short*)(ws + 7 * MB);    // 1 MB
  unsigned short* Kb  = (unsigned short*)(ws + 8 * MB);    // 32 MB
  unsigned short* Qb  = (unsigned short*)(ws + 40 * MB);   // 32 MB
  unsigned short* Vb  = (unsigned short*)(ws + 72 * MB);   // 32 MB
  unsigned short* Ob  = (unsigned short*)(ws + 104 * MB);  // 32 MB  (136 MB total)

  // 1) conversions
  transpose_in_kernel<<<(NB * DD * PP) / 256, 256, 0, stream>>>(x, inT);
  cvt_bf16_kernel<<<(HD * DD) / 256, 256, 0, stream>>>(Wk, WkB, HD * DD);
  cvt_bf16_kernel<<<(HD * DD) / 256, 256, 0, stream>>>(Wq, WqB, HD * DD);
  cvt_bf16_kernel<<<(HD * DD) / 256, 256, 0, stream>>>(Wv, WvB, HD * DD);
  cvt_bf16_kernel<<<(DD * HD) / 256, 256, 0, stream>>>(Wo, WoB, DD * HD);

  // 2) projections
  proj_kq_kernel<<<4096, 128, 0, stream>>>(WkB, bk, inT, Kb);
  proj_kq_kernel<<<4096, 128, 0, stream>>>(WqB, bq, inT, Qb);
  proj_v_kernel<<<4096, 128, 0, stream>>>(WvB, bv, inT, Vb);

  // 3) fused attention (scores + key-axis softmax + V*P)
  attn_kernel<<<NB * HH * (PP / 16), 128, 0, stream>>>(Kb, Qb, Vb, pos, Ob);

  // 4) output projection
  out_proj_kernel<<<512, 128, 0, stream>>>(WoB, bo, Ob, (float*)d_out);
}
